// Dcls2_1d_12713103196287
// MI455X (gfx1250) — matrix operand order repaired and checked by reading, not compiled, not run on hardware
//
#include <hip/hip_runtime.h>

typedef __attribute__((ext_vector_type(16))) __bf16 v16bf;
typedef __attribute__((ext_vector_type(8)))  float  v8f;

typedef unsigned short u16;
typedef unsigned int   u32;
typedef unsigned long long u64;

union FragBF {
    uint4 q[2];
    v16bf v;
};

__device__ __forceinline__ u16 f2bf(float f) {
    u32 u = __float_as_uint(f);
    u32 r = u + 0x7FFFu + ((u >> 16) & 1u);   // round-to-nearest-even
    return (u16)(r >> 16);
}

// gfx1250 async global->LDS copy (ASYNCcnt); GV addressing mode.
__device__ __forceinline__ void async_copy_b128(unsigned ldsAddr, const void* gptr) {
    asm volatile("global_load_async_to_lds_b128 %0, %1, off"
                 :: "v"(ldsAddr), "v"((u64)(uintptr_t)gptr)
                 : "memory");
}
__device__ __forceinline__ void wait_async0() {
    asm volatile("s_wait_asynccnt 0x0" ::: "memory");
}

// ---------------------------------------------------------------------------
// Kernel 1a: x (fp32) -> xb (bf16), vectorized 4-wide. Done once, so the
// cvt VALU work leaves the hot conv loop entirely.
// ---------------------------------------------------------------------------
__global__ void __launch_bounds__(256) x_to_bf16_kernel(
    const float* __restrict__ x, u16* __restrict__ xb, int n4)
{
    int t = blockIdx.x * 256 + threadIdx.x;
    if (t >= n4) return;
    float4 f = ((const float4*)x)[t];
    ushort4 o;
    o.x = f2bf(f.x); o.y = f2bf(f.y); o.z = f2bf(f.z); o.w = f2bf(f.w);
    ((ushort4*)xb)[t] = o;
}

// ---------------------------------------------------------------------------
// Kernel 1b: bilinear-scatter 3x3 weights at learnable row offsets P into the
// dense 7x3 kernel, bf16, laid out Kb[tap=kh'*3+kw][oc=256][ci=128].
// ---------------------------------------------------------------------------
__global__ void __launch_bounds__(256) build_kb_kernel(
    const float* __restrict__ weight, const float* __restrict__ P,
    u16* __restrict__ Kb)
{
    int t = blockIdx.x * blockDim.x + threadIdx.x;   // (oc,ci,kw)
    if (t >= 256 * 128 * 3) return;
    int kw = t % 3;
    int ci = (t / 3) % 128;
    int oc = t / (3 * 128);

    float col[7] = {0.f, 0.f, 0.f, 0.f, 0.f, 0.f, 0.f};
    #pragma unroll
    for (int kh = 0; kh < 3; ++kh) {
        int idx = ((oc * 128 + ci) * 3 + kh) * 3 + kw;
        float wv = weight[idx];
        float p  = fminf(fmaxf(P[idx], -2.f), 2.f);
        float pos  = (float)kh + 2.f + p;            // in [0,6]
        float p0   = floorf(pos);
        float frac = pos - p0;
        int   i0   = (int)p0;
        col[i0] += wv * (1.f - frac);
        if (i0 + 1 < 7) col[i0 + 1] += wv * frac;
    }
    #pragma unroll
    for (int r = 0; r < 7; ++r)
        Kb[(((size_t)(r * 3 + kw)) * 256 + oc) * 128 + ci] = f2bf(col[r]);
}

// ---------------------------------------------------------------------------
// Kernel 2: implicit-GEMM conv, v_wmma_f32_16x16x32_bf16.
// Block: 64 pixels (one w-row at fixed n,h) x 64 oc; 8 wave32s (4x2).
// Per kh: stage 66x128 bf16 A strip ONCE (shared by all kw via row shift and
// all ci-chunks), async-DMA the 64x128 B tap tiles double-buffered.
// ---------------------------------------------------------------------------
#define ASTRIDE 136   // 128 ci + 8 pad halfwords (272B rows, 16B aligned)

__global__ void __launch_bounds__(256) dcls_conv_kernel(
    const u16*   __restrict__ xb,     // (16,128,64,64) bf16
    const u16*   __restrict__ Kb,     // (21,256,128)   bf16
    const float* __restrict__ bias,   // (256,)
    float*       __restrict__ out)    // (16,256,64,64) f32
{
    __shared__ alignas(16) u16 ldsA[66 * ASTRIDE];      // [ww+1][ci]
    __shared__ alignas(16) u16 ldsB[2][64 * ASTRIDE];   // [buf][oc][ci]

    const int ocBase = blockIdx.x * 64;
    const int h      = blockIdx.y;
    const int n      = blockIdx.z;
    const int tid    = threadIdx.x;
    const int lane   = tid & 31;
    const int wave   = tid >> 5;
    const int wm     = wave >> 1;          // pixel quarter
    const int wn     = wave & 1;           // oc half
    const int lrow   = lane & 15;
    const int g      = lane >> 4;          // lane-half (K split)

    // zero-pad columns ww=-1 and ww=64 (rows 0 and 65); never overwritten
    if (tid < 128) {
        ldsA[tid] = 0;
        ldsA[65 * ASTRIDE + tid] = 0;
    }

    v8f acc0 = {};
    v8f acc1 = {};

    int curBuf = 0;
    for (int kh = 0; kh < 7; ++kh) {
        const int hh = h + kh - 3;                 // PAD_H = 3
        if (hh < 0 || hh > 63) continue;           // block-uniform skip

        __syncthreads();   // everyone done reading ldsA / both ldsB buffers

        // ---- kick off async DMA of B tap (kh, kw=0) into curBuf ----
        {
            const u16* tap = Kb + ((size_t)(kh * 3 + 0) * 256 + ocBase) * 128;
            #pragma unroll
            for (int it = 0; it < 4; ++it) {
                int c  = tid + it * 256;           // 1024 x 16B chunks
                int r  = c >> 4;                   // oc row
                int cq = (c & 15) * 8;             // ci halfword offset
                async_copy_b128((unsigned)(uintptr_t)&ldsB[curBuf][r * ASTRIDE + cq],
                                tap + (size_t)r * 128 + cq);
            }
        }

        // ---- stage A strip (transpose w<->ci) while the DMA flies ----
        #pragma unroll
        for (int it = 0; it < 8; ++it) {
            int c  = tid + it * 256;               // 2048 x 4-elem chunks
            int ci = c >> 4;                       // 0..127
            int wq = (c & 15) * 4;                 // 0..60
            ushort4 v = *(const ushort4*)(xb + (((size_t)n * 128 + ci) * 64 + hh) * 64 + wq);
            ldsA[(wq + 1) * ASTRIDE + ci] = v.x;
            ldsA[(wq + 2) * ASTRIDE + ci] = v.y;
            ldsA[(wq + 3) * ASTRIDE + ci] = v.z;
            ldsA[(wq + 4) * ASTRIDE + ci] = v.w;
        }
        // warm next h-row of xb in L2 (global_prefetch)
        if (tid < 128 && hh < 63)
            __builtin_prefetch(xb + (((size_t)n * 128 + tid) * 64 + hh + 1) * 64, 0, 2);

        for (int kw = 0; kw < 3; ++kw) {
            wait_async0();       // curBuf B tile landed in LDS
            __syncthreads();     // publish A (+B); all waves past old reads

            int nextBuf = curBuf ^ 1;
            if (kw < 2) {        // overlap next tap's DMA with this tap's WMMAs
                const u16* tap = Kb + ((size_t)(kh * 3 + kw + 1) * 256 + ocBase) * 128;
                #pragma unroll
                for (int it = 0; it < 4; ++it) {
                    int c  = tid + it * 256;
                    int r  = c >> 4;
                    int cq = (c & 15) * 8;
                    async_copy_b128((unsigned)(uintptr_t)&ldsB[nextBuf][r * ASTRIDE + cq],
                                    tap + (size_t)r * 128 + cq);
                }
            }

            // ---- 8 WMMAs per wave: 4 ci-chunks x 2 oc sub-tiles ----
            const u16* bbuf = ldsB[curBuf];
            #pragma unroll
            for (int cc = 0; cc < 4; ++cc) {
                const int ccB = cc * 32;
                FragBF a, b0, b1;
                const u16* arow = ldsA + (kw + wm * 16 + lrow) * ASTRIDE + ccB + g * 16;
                a.q[0] = *(const uint4*)(arow);        // K 0..7 / 16..23
                a.q[1] = *(const uint4*)(arow + 8);    // K 8..15 / 24..31

                const u16* br0 = bbuf + (wn * 32 + lrow) * ASTRIDE + ccB + g * 16;
                b0.q[0] = *(const uint4*)(br0);
                b0.q[1] = *(const uint4*)(br0 + 8);

                const u16* br1 = bbuf + (wn * 32 + 16 + lrow) * ASTRIDE + ccB + g * 16;
                b1.q[0] = *(const uint4*)(br1);
                b1.q[1] = *(const uint4*)(br1 + 8);

                acc0 = __builtin_amdgcn_wmma_f32_16x16x32_bf16(
                           false, a.v, false, b0.v, (short)0, acc0, false, false);
                acc1 = __builtin_amdgcn_wmma_f32_16x16x32_bf16(
                           false, a.v, false, b1.v, (short)0, acc1, false, false);
            }
            curBuf = nextBuf;
        }
    }

    // ---- epilogue: C layout VGPR i -> (M = i + 8*g, N = lrow) ----
    const int oc0 = ocBase + wn * 32 + lrow;
    const float bv0 = bias[oc0];
    const float bv1 = bias[oc0 + 16];
    #pragma unroll
    for (int i = 0; i < 8; ++i) {
        int wo = wm * 16 + g * 8 + i;
        out[(((size_t)n * 256 + oc0     ) * 64 + h) * 64 + wo] = acc0[i] + bv0;
        out[(((size_t)n * 256 + oc0 + 16) * 64 + h) * 64 + wo] = acc1[i] + bv1;
    }
}

// ---------------------------------------------------------------------------
extern "C" void kernel_launch(void* const* d_in, const int* in_sizes, int n_in,
                              void* d_out, int out_size, void* d_ws, size_t ws_size,
                              hipStream_t stream) {
    const float* x      = (const float*)d_in[0];   // (16,128,64,64)
    const float* weight = (const float*)d_in[1];   // (256,128,3,3)
    const float* bias   = (const float*)d_in[2];   // (256,)
    const float* P      = (const float*)d_in[3];   // (1,256,128,3,3)
    float*       out    = (float*)d_out;

    const size_t KB_BYTES = (size_t)21 * 256 * 128 * 2;   // 1,376,256 (16B-mult)
    u16* Kb = (u16*)d_ws;
    u16* xb = (u16*)((char*)d_ws + KB_BYTES);             // 16.78 MB

    // 1a: x -> bf16 (4-wide)
    int n4 = 16 * 128 * 64 * 64 / 4;
    x_to_bf16_kernel<<<(n4 + 255) / 256, 256, 0, stream>>>(x, xb, n4);

    // 1b: dense bf16 kernel from learnable spacings
    int nthreads = 256 * 128 * 3;
    build_kb_kernel<<<(nthreads + 255) / 256, 256, 0, stream>>>(weight, P, Kb);

    // 2: implicit GEMM conv (WMMA bf16 -> f32, async B staging)
    dim3 grid(4, 64, 16);   // oc-blocks, h, n
    dcls_conv_kernel<<<grid, 256, 0, stream>>>(xb, Kb, bias, out);
}
